// SOMVAE_24824910971535
// MI455X (gfx1250) — compile-verified, matrix operand order checked by reading, not run on hardware
//
#include <hip/hip_runtime.h>
#include <hip/hip_bf16.h>

// ---------------------------------------------------------------------------
// SOM-VAE forward on MI455X (gfx1250, wave32, WMMA).
// Roofline: ~860 MB moved @23.3TB/s ~= 37us; 17 GFLOP of real GEMM work done
// with v_wmma_f32_16x16x32_bf16 so matrix math is never the bottleneck.
// z_q @ W_dec_q is replaced by a 64-row precomputed table + gather.
// ---------------------------------------------------------------------------

typedef __attribute__((ext_vector_type(16))) __bf16 v16bf;  // 32B: WMMA A/B frag
typedef __attribute__((ext_vector_type(8)))  __bf16 v8bf;   // 16B
typedef __attribute__((ext_vector_type(8)))  float  v8f;    // WMMA C/D frag
typedef __attribute__((ext_vector_type(4)))  float  f4;     // 16B vector ld/st

#define SOM_B      32768
#define SOM_D      2048
#define SOM_L      64
#define SOM_CODES  64

// ---------------------------------------------------------------------------
// Kernel 0: pack W_enc [2048,64] and W_dec_e [64,2048] into bf16 WMMA
// B-fragment order:  frag[ntile][kchunk][lane][16 bf16].
// bf16 A/B lane layout (ISA 7.12.2): lane half kg=L>>4 selects K offsets
//   e=0..7  -> K = 8*kg + e ;  e=8..15 -> K = 16 + 8*kg + (e-8)
// ---------------------------------------------------------------------------
__global__ __launch_bounds__(256) void pack_weights(
    const float* __restrict__ W_enc, const float* __restrict__ W_dec_e,
    __bf16* __restrict__ fragWenc, __bf16* __restrict__ fragWdec) {
  int t = blockIdx.x * blockDim.x + threadIdx.x;  // 0..16383
  if (t < 8192) {
    // W_enc: 4 ntiles x 64 kchunks x 32 lanes
    int lane  = t & 31;
    int chunk = (t >> 5) & 63;
    int ntile = t >> 11;
    int n  = ntile * 16 + (lane & 15);
    int kg = lane >> 4;
    int kbase = chunk * 32 + 8 * kg;
    __bf16* dst = fragWenc + (size_t)t * 16;
#pragma unroll
    for (int e = 0; e < 16; ++e) {
      int k = kbase + ((e >= 8) ? 16 : 0) + (e & 7);
      dst[e] = (__bf16)W_enc[k * SOM_L + n];
    }
  } else {
    // W_dec_e: 128 ntiles x 2 kchunks x 32 lanes
    int u = t - 8192;
    int lane  = u & 31;
    int chunk = (u >> 5) & 1;
    int ntile = u >> 6;
    int n  = ntile * 16 + (lane & 15);
    int kg = lane >> 4;
    int kbase = chunk * 32 + 8 * kg;
    __bf16* dst = fragWdec + (size_t)u * 16;
#pragma unroll
    for (int e = 0; e < 16; ++e) {
      int k = kbase + ((e >= 8) ? 16 : 0) + (e & 7);
      dst[e] = (__bf16)W_dec_e[k * SOM_D + n];
    }
  }
}

// ---------------------------------------------------------------------------
// Kernel 1: z_e = x @ W_enc + b_enc   [32768,2048]x[2048,64]
// 8 waves/block: wave -> (mtile 0..1, ntile 0..3); block covers 32 rows x 64.
// K loop: 64 chunks of 32; A = fp32 x converted to bf16 in-flight (b128 loads),
// B = packed frags from L2-resident 256KB table.
// ---------------------------------------------------------------------------
__global__ __launch_bounds__(256) void encoder_wmma(
    const float* __restrict__ x, const float* __restrict__ b_enc,
    const __bf16* __restrict__ fragWenc,
    float* __restrict__ z_e_out, __bf16* __restrict__ z_e_bf) {
  const int lane  = threadIdx.x & 31;
  const int wave  = threadIdx.x >> 5;
  const int mtile = wave & 1;
  const int ntile = wave >> 1;
  const int mbase = blockIdx.x * 32 + mtile * 16;
  const int kg    = lane >> 4;
  const int l15   = lane & 15;
  const float* __restrict__ xrow = x + (size_t)(mbase + l15) * SOM_D;
  const v16bf* __restrict__ bfr =
      (const v16bf*)fragWenc + ((size_t)ntile * 64 * 32 + lane);

  v8f c = {};
  for (int ch = 0; ch < 64; ++ch) {
    const int k0 = ch * 32 + 8 * kg;
    const f4* xr4 = (const f4*)(xrow + k0);     // 32B-aligned
    f4 p0 = xr4[0], p1 = xr4[1];                // K = k0 .. k0+7
    f4 q0 = xr4[4], q1 = xr4[5];                // K = k0+16 .. k0+23
    v16bf a;
#pragma unroll
    for (int e = 0; e < 4; ++e) {
      a[e]      = (__bf16)p0[e];
      a[4 + e]  = (__bf16)p1[e];
      a[8 + e]  = (__bf16)q0[e];
      a[12 + e] = (__bf16)q1[e];
    }
    v16bf b = bfr[ch * 32];
    c = __builtin_amdgcn_wmma_f32_16x16x32_bf16(false, a, false, b,
                                                (short)0, c, false, false);
  }
  const int n = ntile * 16 + l15;               // C/D: lane -> column
  const float bias = b_enc[n];
#pragma unroll
  for (int v = 0; v < 8; ++v) {                 // C/D: vgpr v -> row v + 8*kg
    const int mm = mbase + v + 8 * kg;
    const float val = c[v] + bias;
    z_e_out[(size_t)mm * SOM_L + n] = val;
    z_e_bf [(size_t)mm * SOM_L + n] = (__bf16)val;
  }
}

// ---------------------------------------------------------------------------
// Kernel 2: distances + argmin. One wave32 per row; lane j handles codes
// j and j+32; wave-wide shfl_xor butterfly argmin (first-min tiebreak).
// ---------------------------------------------------------------------------
__global__ __launch_bounds__(256) void dist_argmin(
    const float* __restrict__ z_e, const float* __restrict__ emb,
    float* __restrict__ z_dist, float* __restrict__ k_f, int* __restrict__ k_i) {
  __shared__ float Es[SOM_CODES * SOM_L];
  __shared__ float ee[SOM_CODES];
  for (int i = threadIdx.x; i < SOM_CODES * SOM_L; i += 256) Es[i] = emb[i];
  __syncthreads();
  if (threadIdx.x < SOM_CODES) {
    float s = 0.f;
    for (int l = 0; l < SOM_L; ++l) { float v = Es[threadIdx.x * SOM_L + l]; s += v * v; }
    ee[threadIdx.x] = s;
  }
  __syncthreads();

  const int lane = threadIdx.x & 31;
  const int row  = blockIdx.x * 8 + (threadIdx.x >> 5);
  const float* __restrict__ zr = z_e + (size_t)row * SOM_L;

  float r[SOM_L];
#pragma unroll
  for (int l = 0; l < SOM_L; ++l) r[l] = zr[l];
  float sq = 0.f;
#pragma unroll
  for (int l = 0; l < SOM_L; ++l) sq += r[l] * r[l];

  float dot0 = 0.f, dot1 = 0.f;
#pragma unroll
  for (int l = 0; l < SOM_L; ++l) dot0 += r[l] * Es[lane * SOM_L + l];
#pragma unroll
  for (int l = 0; l < SOM_L; ++l) dot1 += r[l] * Es[(lane + 32) * SOM_L + l];
  const float d0 = sq - 2.f * dot0 + ee[lane];
  const float d1 = sq - 2.f * dot1 + ee[lane + 32];
  z_dist[(size_t)row * 64 + lane]      = d0;
  z_dist[(size_t)row * 64 + lane + 32] = d1;

  float bestd = d0; int besti = lane;
  if (d1 < bestd) { bestd = d1; besti = lane + 32; }
#pragma unroll
  for (int off = 16; off > 0; off >>= 1) {
    float od = __shfl_xor(bestd, off, 32);
    int   oi = __shfl_xor(besti, off, 32);
    if (od < bestd || (od == bestd && oi < besti)) { bestd = od; besti = oi; }
  }
  if (lane == 0) { k_f[row] = (float)besti; k_i[row] = besti; }
}

// ---------------------------------------------------------------------------
// Kernel 3: dec_q table: table[code][n] = E[code]@W_dec_q[:,n] + b  (17 MFLOP)
// ---------------------------------------------------------------------------
__global__ __launch_bounds__(256) void dec_table(
    const float* __restrict__ emb, const float* __restrict__ W_dec_q,
    const float* __restrict__ b_dec_q, float* __restrict__ table) {
  int gid  = blockIdx.x * blockDim.x + threadIdx.x;  // 64*2048
  int n    = gid & (SOM_D - 1);
  int code = gid >> 11;
  float s = b_dec_q[n];
  const float* __restrict__ e = emb + code * SOM_L;
  for (int l = 0; l < SOM_L; ++l) s += e[l] * W_dec_q[l * SOM_D + n];
  table[gid] = s;
}

// ---------------------------------------------------------------------------
// Kernel 4: gather x_q rows from table; z_q + 5 neighbor slices from E.
// Streaming stores, b128-granular. z_q_right stays zero (faithful to ref bug).
// ---------------------------------------------------------------------------
__global__ __launch_bounds__(256) void gather_out(
    const int* __restrict__ ki, const float* __restrict__ table,
    const float* __restrict__ emb,
    float* __restrict__ x_q, float* __restrict__ z_q, float* __restrict__ neigh) {
  int gid = blockIdx.x * blockDim.x + threadIdx.x;  // 32768 * 512
  int c4  = gid & 511;
  int row = gid >> 9;
  int code = ki[row];
  const f4* __restrict__ t4 = (const f4*)(table + (size_t)code * SOM_D);
  ((f4*)(x_q + (size_t)row * SOM_D))[c4] = t4[c4];
  if (c4 < 16) {
    const int k1 = code >> 3, k2 = code & 7;
    const f4* __restrict__ E4 = (const f4*)emb;   // 16 f4 per code row
    const f4 zero = {0.f, 0.f, 0.f, 0.f};
    f4 zq = E4[code * 16 + c4];
    ((f4*)(z_q + (size_t)row * SOM_L))[c4] = zq;
    f4* nb = (f4*)(neigh + (size_t)row * 5 * SOM_L);
    nb[0 * 16 + c4] = zq;
    nb[1 * 16 + c4] = (k1 < 7) ? E4[((k1 + 1) * 8 + k2) * 16 + c4] : zero;
    nb[2 * 16 + c4] = (k1 > 0) ? E4[((k1 - 1) * 8 + k2) * 16 + c4] : zero;
    nb[3 * 16 + c4] = zero;                        // '==' bug in reference
    nb[4 * 16 + c4] = (k2 > 0) ? E4[(k1 * 8 + (k2 - 1)) * 16 + c4] : zero;
  }
}

// ---------------------------------------------------------------------------
// Kernel 5: x_e = z_e @ W_dec_e + b_dec_e   [32768,64]x[64,2048], K=64 -> 2 WMMA
// per 16x16 tile. B frags held in registers across nothing to reuse (K tiny);
// A is bf16 z_e from ws as b128 loads.
// ---------------------------------------------------------------------------
__global__ __launch_bounds__(256) void xe_wmma(
    const __bf16* __restrict__ z_bf, const __bf16* __restrict__ fragW,
    const float* __restrict__ b_dec_e, float* __restrict__ x_e) {
  const int lane  = threadIdx.x & 31;
  const int wave  = threadIdx.x >> 5;
  const int ntile = blockIdx.x;                  // 0..127
  const int mtile = blockIdx.y * 8 + wave;       // 0..2047
  const int mbase = mtile * 16;
  const int kg    = lane >> 4;
  const int l15   = lane & 15;
  const __bf16* __restrict__ zr = z_bf + (size_t)(mbase + l15) * SOM_L;

  const v16bf* __restrict__ bf = (const v16bf*)fragW + ((size_t)ntile * 64 + lane);
  v16bf b0 = bf[0];
  v16bf b1 = bf[32];

  v8f c = {};
#pragma unroll
  for (int ch = 0; ch < 2; ++ch) {
    const int k0 = ch * 32 + 8 * kg;
    const v8bf* p = (const v8bf*)(zr + k0);      // 16B aligned
    v8bf lo = p[0];
    v8bf hi = p[2];                              // +16 bf16 elements
    v16bf a;
#pragma unroll
    for (int e = 0; e < 8; ++e) { a[e] = lo[e]; a[8 + e] = hi[e]; }
    c = __builtin_amdgcn_wmma_f32_16x16x32_bf16(false, a, false,
                                                ch ? b1 : b0,
                                                (short)0, c, false, false);
  }
  const int n = ntile * 16 + l15;
  const float bias = b_dec_e[n];
#pragma unroll
  for (int v = 0; v < 8; ++v)
    x_e[(size_t)(mbase + v + 8 * kg) * SOM_D + n] = c[v] + bias;
}

// ---------------------------------------------------------------------------
extern "C" void kernel_launch(void* const* d_in, const int* in_sizes, int n_in,
                              void* d_out, int out_size, void* d_ws, size_t ws_size,
                              hipStream_t stream) {
  (void)in_sizes; (void)n_in; (void)out_size; (void)ws_size;
  const float* x       = (const float*)d_in[0];
  const float* W_enc   = (const float*)d_in[1];
  const float* b_enc   = (const float*)d_in[2];
  const float* W_dec_q = (const float*)d_in[3];
  const float* b_dec_q = (const float*)d_in[4];
  const float* W_dec_e = (const float*)d_in[5];
  const float* b_dec_e = (const float*)d_in[6];
  const float* emb     = (const float*)d_in[7];

  float* out = (float*)d_out;   // tuple concat, all fp32 in return order
  float* x_e    = out;                                   // 32768*2048
  float* x_q    = out + (size_t)67108864;                // 32768*2048
  float* z_e    = out + (size_t)134217728;               // 32768*64
  float* z_q    = out + (size_t)136314880;               // 32768*64
  float* neigh  = out + (size_t)138412032;               // 32768*5*64
  float* k_f    = out + (size_t)148897792;               // 32768
  float* z_dist = out + (size_t)148930560;               // 32768*64

  char* ws = (char*)d_ws;
  __bf16* z_bf     = (__bf16*)(ws);                      // 4 MB
  __bf16* fragWenc = (__bf16*)(ws + 4194304);            // 256 KB
  __bf16* fragWdec = (__bf16*)(ws + 4456448);            // 256 KB
  float*  table    = (float*) (ws + 4718592);            // 512 KB
  int*    k_i      = (int*)   (ws + 5242880);            // 128 KB

  pack_weights<<<64,    256, 0, stream>>>(W_enc, W_dec_e, fragWenc, fragWdec);
  encoder_wmma<<<1024,  256, 0, stream>>>(x, b_enc, fragWenc, z_e, z_bf);
  dist_argmin <<<4096,  256, 0, stream>>>(z_e, emb, z_dist, k_f, k_i);
  dec_table   <<<512,   256, 0, stream>>>(emb, W_dec_q, b_dec_q, table);
  gather_out  <<<65536, 256, 0, stream>>>(k_i, table, emb, x_q, z_q, neigh);
  xe_wmma     <<<dim3(128, 256), 256, 0, stream>>>(z_bf, fragWdec, b_dec_e, x_e);
}
